// StructureEmbedding_19396072309454
// MI455X (gfx1250) — matrix operand order repaired and checked by reading, not compiled, not run on hardware
//
#include <hip/hip_runtime.h>
#include <hip/hip_bf16.h>

#define BB 2
#define LL 2048
#define KK 48
#define NUM_RBF 16
#define NUM_POS 16
#define MAX_REL 32
#define EDGE 128
#define N_ATOMS 5
#define EDGE_IN (NUM_POS + N_ATOMS * N_ATOMS * NUM_RBF)   // 416
#define KT (EDGE_IN / 32)                                  // 13 k-tiles of 32

typedef __attribute__((ext_vector_type(16))) __bf16 v16bf;
typedef __attribute__((ext_vector_type(8)))  float  v8f;

// ---------- helpers ----------
__device__ inline unsigned short f32_to_bf16(float f) {
    union { float f; unsigned u; } x; x.f = f;
    unsigned u = x.u;
    u += 0x7FFFu + ((u >> 16) & 1u);   // round to nearest even
    return (unsigned short)(u >> 16);
}

// A fragment (16-bit A, 16x32, wave32) from LDS feature tile, row stride 416.
// Lane l: M = l&15; kb = (l<16)?0:8; VGPR v: K = (v>=4?16:0)+kb+2*(v&3), +1
__device__ inline v16bf load_frag_A(const unsigned short* F, int mt, int kt, int lane) {
    int M  = mt * 16 + (lane & 15);
    int kb = (lane < 16) ? 0 : 8;
    const unsigned short* row = F + M * EDGE_IN + kt * 32;
    union { unsigned u[8]; v16bf v; } r;
#pragma unroll
    for (int v = 0; v < 8; ++v) {
        int kk = ((v & 4) ? 16 : 0) + kb + 2 * (v & 3);
        r.u[v] = *(const unsigned*)(row + kk);
    }
    return r.v;
}

// B fragment (16-bit B, 32x16, wave32) from global W^T stored [N=128][K=416].
// Lane l: N = l&15; kb = (l<16)?0:8; VGPR v: K = (v>=4?16:0)+kb+2*(v&3), +1
__device__ inline v16bf load_frag_B(const unsigned short* Wt, int nt, int kt, int lane) {
    int N  = nt * 16 + (lane & 15);
    int kb = (lane < 16) ? 0 : 8;
    const unsigned short* row = Wt + N * EDGE_IN + kt * 32;
    union { unsigned u[8]; v16bf v; } r;
#pragma unroll
    for (int v = 0; v < 8; ++v) r.u[v] = *(const unsigned*)(row + ((v & 4) ? 16 : 0) + kb + 2 * (v & 3));
    return r.v;
}

// ---------- kernel 1: KNN (one block of 128 threads per residue) ----------
__global__ void knn_kernel(const float* __restrict__ X, const float* __restrict__ mask,
                           int* __restrict__ eidx_out) {
    __shared__ float Dbuf[LL];
    __shared__ float sv[128];
    __shared__ int   si[128];

    const int bi = blockIdx.x;          // b*L + i
    const int b  = bi / LL;
    const int i  = bi % LL;
    const int tid = threadIdx.x;

    const float cax = X[((size_t)bi * 4 + 1) * 3 + 0];
    const float cay = X[((size_t)bi * 4 + 1) * 3 + 1];
    const float caz = X[((size_t)bi * 4 + 1) * 3 + 2];
    const float mi  = mask[bi];

    // pass 1: masked distances + row max
    float lmax = -1e30f;
    for (int j = tid; j < LL; j += 128) {
        const size_t jb = (size_t)(b * LL + j);
        float dx = X[(jb * 4 + 1) * 3 + 0] - cax;
        float dy = X[(jb * 4 + 1) * 3 + 1] - cay;
        float dz = X[(jb * 4 + 1) * 3 + 2] - caz;
        float m2 = mi * mask[jb];
        float d  = m2 * sqrtf(dx * dx + dy * dy + dz * dz + 1e-6f);
        Dbuf[j] = d;
        lmax = fmaxf(lmax, d);
    }
    sv[tid] = lmax;
    __syncthreads();
    for (int s = 64; s > 0; s >>= 1) {
        if (tid < s) sv[tid] = fmaxf(sv[tid], sv[tid + s]);
        __syncthreads();
    }
    const float Dmax = sv[0];
    __syncthreads();

    // pass 2: D_adjust = D + (1-m2)*Dmax
    for (int j = tid; j < LL; j += 128) {
        float m2 = mi * mask[(size_t)(b * LL + j)];
        Dbuf[j] += (1.0f - m2) * Dmax;
    }
    __syncthreads();

    // selection: 48 x argmin with stable tie-break (smaller index wins)
    for (int sel = 0; sel < KK; ++sel) {
        float bv = __builtin_inff();
        int   bj = -1;
        for (int j = tid; j < LL; j += 128) {
            float v = Dbuf[j];
            if (v < bv || (v == bv && j < bj)) { bv = v; bj = j; }
        }
        sv[tid] = bv; si[tid] = bj;
        __syncthreads();
        for (int s = 64; s > 0; s >>= 1) {
            if (tid < s) {
                float vo = sv[tid + s]; int io = si[tid + s];
                if (vo < sv[tid] || (vo == sv[tid] && io < si[tid])) { sv[tid] = vo; si[tid] = io; }
            }
            __syncthreads();
        }
        if (tid == 0) {
            eidx_out[(size_t)bi * KK + sel] = si[0];
            Dbuf[si[0]] = __builtin_inff();
        }
        __syncthreads();
    }
}

// ---------- kernel 2: per-residue atoms [Ca, N, C, O, Cb] -> ws ----------
__global__ void atoms_kernel(const float* __restrict__ X, float* __restrict__ atoms) {
    const int t = blockIdx.x * blockDim.x + threadIdx.x;
    if (t >= BB * LL) return;
    const float* r = X + (size_t)t * 12;
    float Nx = r[0], Ny = r[1], Nz = r[2];
    float Cax = r[3], Cay = r[4], Caz = r[5];
    float Cx = r[6], Cy = r[7], Cz = r[8];
    float Ox = r[9], Oy = r[10], Oz = r[11];
    float bx = Cax - Nx, by = Cay - Ny, bz = Caz - Nz;   // bb
    float cx = Cx - Cax, cy = Cy - Cay, cz = Cz - Caz;   // cc
    float ax = by * cz - bz * cy;                        // aa = cross(bb, cc)
    float ay = bz * cx - bx * cz;
    float az = bx * cy - by * cx;
    float Cbx = -0.58273431f * ax + 0.56802827f * bx - 0.54067466f * cx + Cax;
    float Cby = -0.58273431f * ay + 0.56802827f * by - 0.54067466f * cy + Cay;
    float Cbz = -0.58273431f * az + 0.56802827f * bz - 0.54067466f * cz + Caz;
    float* o = atoms + (size_t)t * 15;
    o[0] = Cax; o[1] = Cay; o[2] = Caz;
    o[3] = Nx;  o[4] = Ny;  o[5] = Nz;
    o[6] = Cx;  o[7] = Cy;  o[8] = Cz;
    o[9] = Ox;  o[10] = Oy; o[11] = Oz;
    o[12] = Cbx; o[13] = Cby; o[14] = Cbz;
}

// ---------- kernel 3: W_edge (416x128 f32, row-major K x N) -> bf16 W^T [N][K] ----------
__global__ void wconv_kernel(const float* __restrict__ W, unsigned short* __restrict__ Wt) {
    const int t = blockIdx.x * blockDim.x + threadIdx.x;
    if (t >= EDGE_IN * EDGE) return;
    const int k = t / EDGE;
    const int n = t % EDGE;
    Wt[(size_t)n * EDGE_IN + k] = f32_to_bf16(W[t]);
}

// ---------- kernel 4: fused features + WMMA GEMM + LayerNorm ----------
// one block (256 threads = 8 waves) per residue: 48 edges x 416 feats -> 48 x 128
__global__ void edge_kernel(const float* __restrict__ atoms,
                            const int* __restrict__ ridx, const int* __restrict__ chain,
                            const float* __restrict__ W_pos, const float* __restrict__ b_pos,
                            const unsigned short* __restrict__ Wt,
                            const float* __restrict__ gamma, const float* __restrict__ beta,
                            const int* __restrict__ eidx, float* __restrict__ out) {
    __shared__ unsigned short F[KK * EDGE_IN];   // 48x416 bf16 features (39,936 B)
    __shared__ float Els[KK * EDGE];             // raw GEMM result (24,576 B)
    __shared__ int   sidx[KK];
    __shared__ float smean[KK];
    __shared__ float srstd[KK];

    const int bi = blockIdx.x;                   // b*L + i
    const int b  = bi / LL;
    const int i  = bi % LL;
    const int tid  = threadIdx.x;
    const int lane = tid & 31;
    const int wave = tid >> 5;

    if (tid < KK) sidx[tid] = eidx[(size_t)bi * KK + tid];
    // warm the L2/L0 path for W^T (emits global_prefetch_b8)
    __builtin_prefetch(Wt + (size_t)(tid & 127) * EDGE_IN, 0, 1);
    __syncthreads();

    // positional features: columns [0,16)
    if (tid < KK) {
        const int j    = sidx[tid];
        const int off  = ridx[bi] - ridx[b * LL + j];
        const int same = (chain[bi] == chain[b * LL + j]) ? 1 : 0;
        int d = off + MAX_REL;
        d = d < 0 ? 0 : (d > 2 * MAX_REL ? 2 * MAX_REL : d);
        d = same ? d : (2 * MAX_REL + 1);
        for (int c = 0; c < NUM_POS; ++c)
            F[tid * EDGE_IN + c] = f32_to_bf16(W_pos[d * NUM_POS + c] + b_pos[c]);
    }

    // RBF features: 48 edges x 25 atom pairs, columns [16, 416)
    const float* Ai = atoms + (size_t)bi * 15;
    for (int item = tid; item < KK * 25; item += 256) {
        const int kk = item / 25;
        const int p  = item % 25;
        const int as = p / 5;          // self atom
        const int an = p % 5;          // neighbor atom
        const int j  = sidx[kk];
        const float* Aj = atoms + (size_t)(b * LL + j) * 15;
        float dx = Ai[as * 3 + 0] - Aj[an * 3 + 0];
        float dy = Ai[as * 3 + 1] - Aj[an * 3 + 1];
        float dz = Ai[as * 3 + 2] - Aj[an * 3 + 2];
        float dp = sqrtf(dx * dx + dy * dy + dz * dz + 1e-6f);
        unsigned short* dst = F + kk * EDGE_IN + NUM_POS + p * NUM_RBF;
#pragma unroll
        for (int r = 0; r < NUM_RBF; ++r) {
            float mu = 2.0f + (float)r * (20.0f / 15.0f);
            float t  = (dp - mu) * (1.0f / 1.25f);
            dst[r] = f32_to_bf16(__expf(-t * t));
        }
    }
    __syncthreads();

    // GEMM: wave w owns n-tile w; loops 3 m-tiles x 13 k-tiles of wmma_f32_16x16x32_bf16
    const int nt = wave;
#pragma unroll
    for (int mt = 0; mt < 3; ++mt) {
        v8f acc = {0.f, 0.f, 0.f, 0.f, 0.f, 0.f, 0.f, 0.f};
        for (int kt = 0; kt < KT; ++kt) {
            v16bf a  = load_frag_A(F, mt, kt, lane);
            v16bf bf = load_frag_B(Wt, nt, kt, lane);
            acc = __builtin_amdgcn_wmma_f32_16x16x32_bf16(
                false, a, false, bf, (short)0, acc, false, false);
        }
        // C/D layout: VGPR r -> row mt*16 + r + (lane<16?0:8); col nt*16 + (lane&15)
        const int col = nt * 16 + (lane & 15);
        const int rb  = mt * 16 + ((lane < 16) ? 0 : 8);
#pragma unroll
        for (int r = 0; r < 8; ++r) Els[(rb + r) * EDGE + col] = acc[r];
    }
    __syncthreads();

    // LayerNorm over 128 per edge row
    if (tid < KK) {
        const float* row = Els + tid * EDGE;
        float s = 0.f;
        for (int c = 0; c < EDGE; ++c) s += row[c];
        const float m = s * (1.0f / EDGE);
        float v = 0.f;
        for (int c = 0; c < EDGE; ++c) { float d = row[c] - m; v += d * d; }
        smean[tid] = m;
        srstd[tid] = rsqrtf(v * (1.0f / EDGE) + 1e-5f);
    }
    __syncthreads();

    float* orow = out + (size_t)bi * KK * EDGE;
    for (int idx = tid; idx < KK * EDGE; idx += 256) {
        const int r = idx / EDGE;
        const int c = idx % EDGE;
        orow[idx] = (Els[idx] - smean[r]) * srstd[r] * gamma[c] + beta[c];
    }
}

// ---------- launch ----------
extern "C" void kernel_launch(void* const* d_in, const int* in_sizes, int n_in,
                              void* d_out, int out_size, void* d_ws, size_t ws_size,
                              hipStream_t stream) {
    (void)in_sizes; (void)n_in; (void)out_size; (void)ws_size;
    const float* X      = (const float*)d_in[0];
    const float* mask   = (const float*)d_in[1];
    const int*   ridx   = (const int*)d_in[2];
    const int*   chain  = (const int*)d_in[3];
    const float* W_pos  = (const float*)d_in[4];
    const float* b_pos  = (const float*)d_in[5];
    const float* W_edge = (const float*)d_in[6];
    const float* gamma  = (const float*)d_in[7];
    const float* beta   = (const float*)d_in[8];

    float* outE = (float*)d_out;
    int*   outI = (int*)(outE + (size_t)BB * LL * KK * EDGE);  // E_idx tail (int32)

    float*          atoms = (float*)d_ws;                       // B*L*15 f32
    unsigned short* Wt    = (unsigned short*)((char*)d_ws + (size_t)BB * LL * 15 * sizeof(float));

    knn_kernel<<<BB * LL, 128, 0, stream>>>(X, mask, outI);
    atoms_kernel<<<(BB * LL + 255) / 256, 256, 0, stream>>>(X, atoms);
    wconv_kernel<<<(EDGE_IN * EDGE + 255) / 256, 256, 0, stream>>>(W_edge, Wt);
    edge_kernel<<<BB * LL, 256, 0, stream>>>(atoms, ridx, chain, W_pos, b_pos,
                                             Wt, gamma, beta, outI, outE);
}
